// LocalSelectiveSSMLayer_60842506715587
// MI455X (gfx1250) — matrix operand order repaired and checked by reading, not compiled, not run on hardware
//
#include <hip/hip_runtime.h>
#include <hip/hip_bf16.h>
#include <cstdint>

// ---------------------------------------------------------------------------
// Selective SSM (Mamba-style) for MI455X / gfx1250, wave32 + WMMA bf16 + TDM.
//
//   K1 transpose_convert : x[B,L,D] f32 -> x_T[B,D,L] f32  +  x_bf16[BL,D]
//   K2 convert_w         : (Wd|Wb|Wc|0-pad) f32 -> w_bf16[1088,1024]
//   K3 gemm_proj         : WMMA bf16 GEMM [BL,1024]x[1024,1056].
//                          Weights staged to LDS by the Tensor Data Mover
//                          (tensor_load_to_lds, TENSORcnt, double-buffered,
//                          HW row padding for bank-conflict-free ds reads).
//                          Each wave computes 32(m) x 64(e); epilogue does
//                          bias + softplus(delta) and transposed stores.
//   K4 scan_kernel       : strictly-sequential-in-t scan (clip() breaks
//                          associativity); lane=(d,n); chunk=16 steps.
// ---------------------------------------------------------------------------

#define BB 4
#define LL 8192
#define DD 1024
#define NN 16
#define EE (DD + 2 * NN)        // 1056 real output channels
#define EE_PAD 1088             // 17 * 64, zero-padded weight rows
#define KC 128                  // k-elements staged per TDM chunk
#define NKC (DD / KC)           // 8 chunks
#define WROW 272                // LDS row stride: 256B data + 16B pad
#define WBUF (64 * WROW)        // 17408 B per buffer

typedef __bf16 v16bf __attribute__((ext_vector_type(16)));
typedef __bf16 v8bf  __attribute__((ext_vector_type(8)));
typedef float  v8f   __attribute__((ext_vector_type(8)));
typedef unsigned int v4ui __attribute__((ext_vector_type(4)));
typedef int v8i_ __attribute__((ext_vector_type(8)));
typedef int v4i_ __attribute__((ext_vector_type(4)));

#if defined(__has_builtin)
#if __has_builtin(__builtin_amdgcn_tensor_load_to_lds)
#define HAVE_TDM 1
#endif
#endif
#ifndef HAVE_TDM
#define HAVE_TDM 0
#endif

__device__ __forceinline__ unsigned short f2bf(float f) {
    unsigned int u = __float_as_uint(f);
    u += 0x7FFFu + ((u >> 16) & 1u);           // round-to-nearest-even
    return (unsigned short)(u >> 16);
}

__device__ __forceinline__ float softplusf(float z) {
    return (z > 20.0f) ? z : log1pf(__expf(z));
}

#if HAVE_TDM
// One 2D TDM descriptor: 64 rows x 128 k of bf16 weights -> LDS, with HW
// padding of 4 DWORDs every 64 DWORDs (=> 272B LDS row stride).
__device__ __forceinline__ void tdm_stage_weights(unsigned int lds_addr,
                                                  unsigned long long gaddr) {
    v4ui g0 = { 1u,                                   // count=1 valid D#
                lds_addr,
                (unsigned int)(gaddr & 0xFFFFFFFFu),
                (unsigned int)((gaddr >> 32) & 0x1FFFFFFu) | 0x80000000u }; // type=2
    v8i_ g1 = { (int)0x07510000,                      // data_size=2B,pad_en,int=64DW,amt=4DW
                (int)(((unsigned)DD & 0xFFFFu) << 16),// tensor_dim0[15:0]=1024
                (int)((EE_PAD & 0xFFFFu) << 16),      // tensor_dim1[15:0]=1088
                (int)(KC << 16),                      // tile_dim0=128
                (int)64,                              // tile_dim1=64 rows
                (int)DD,                              // tensor_dim0_stride=1024
                0, 0 };
    v4i_ z4 = { 0, 0, 0, 0 };
#if __clang_major__ >= 23
    v8i_ z8 = { 0, 0, 0, 0, 0, 0, 0, 0 };
    __builtin_amdgcn_tensor_load_to_lds(g0, g1, z4, z4, z8, 0);
#else
    __builtin_amdgcn_tensor_load_to_lds(g0, g1, z4, z4, 0);
#endif
}

__device__ __forceinline__ void wait_tensor0() {
#if __has_builtin(__builtin_amdgcn_s_wait_tensorcnt)
    __builtin_amdgcn_s_wait_tensorcnt((short)0);
#else
    asm volatile("s_wait_tensorcnt 0x0" ::: "memory");
#endif
}
#endif

// ---------------------------------------------------------------------------
// K1: x[B,L,D] -> x_T[B,D,L] (f32) and x_bf16[BL,D]
// ---------------------------------------------------------------------------
__global__ __launch_bounds__(256)
void transpose_convert(const float* __restrict__ x,
                       float* __restrict__ xT,
                       unsigned short* __restrict__ xb) {
    __shared__ float tile[32][33];
    const int b  = blockIdx.z;
    const int t0 = blockIdx.x * 32;
    const int d0 = blockIdx.y * 32;
    const int c  = threadIdx.x & 31;
    const int r0 = threadIdx.x >> 5;
    const float* src = x + ((size_t)b * LL + t0) * DD + d0;
#pragma unroll
    for (int i = 0; i < 4; ++i) {
        const int r = r0 + i * 8;
        const float v = src[(size_t)r * DD + c];
        tile[r][c] = v;
        xb[((size_t)(b * LL + t0 + r)) * DD + d0 + c] = f2bf(v);
    }
    __syncthreads();
    float* dstT = xT + ((size_t)b * DD + d0) * LL + t0;
#pragma unroll
    for (int i = 0; i < 4; ++i) {
        const int r = r0 + i * 8;
        dstT[(size_t)r * LL + c] = tile[c][r];
    }
}

// ---------------------------------------------------------------------------
// K2: Wd(1024x1024)|Wb(16x1024)|Wc(16x1024)|zeros -> bf16 [1088,1024]
// ---------------------------------------------------------------------------
__global__ __launch_bounds__(256)
void convert_w(const float* __restrict__ Wd,
               const float* __restrict__ Wb,
               const float* __restrict__ Wc,
               unsigned short* __restrict__ wb) {
    const int idx = blockIdx.x * 256 + threadIdx.x;   // EE_PAD*DD exact
    const int e = idx >> 10;
    const int d = idx & 1023;
    float v;
    if (e < DD)            v = Wd[(size_t)e * DD + d];
    else if (e < DD + NN)  v = Wb[(size_t)(e - DD) * DD + d];
    else if (e < EE)       v = Wc[(size_t)(e - DD - NN) * DD + d];
    else                   v = 0.0f;
    wb[idx] = f2bf(v);
}

// ---------------------------------------------------------------------------
// K3: WMMA bf16 GEMM with TDM-staged weights.
// Block = 8 waves, each wave computes 32(m) x 64(e); block tile 256m x 64e.
// Grid = (17 e-groups, 128 m-supertiles).
// Weight LDS tile: 64 rows x 128 k bf16, 272B row stride (TDM pad) =>
// B-frag ds_load_b128 reads from 16 lanes land on banks 4n..4n+3: no conflict.
// ---------------------------------------------------------------------------
__global__ __launch_bounds__(256)
void gemm_proj(const unsigned short* __restrict__ xb,
               const unsigned short* __restrict__ wb,
               const float* __restrict__ bd,
               const float* __restrict__ bb,
               const float* __restrict__ bc,
               float* __restrict__ deltaT,
               float* __restrict__ BmT,
               float* __restrict__ CmT) {
    __shared__ __align__(16) unsigned char ldsW[2 * WBUF];

    const int lane = threadIdx.x & 31;
    const int wave = threadIdx.x >> 5;
    const int h    = lane >> 4;
    const int n    = lane & 15;

    const int eg    = blockIdx.x;                // 0..16
    const int eBase = eg * 64;
    const int mb    = blockIdx.y * 256 + wave * 32;

    const unsigned short* pa0 = xb + (size_t)(mb + n) * DD + h * 16;
    const unsigned short* pa1 = xb + (size_t)(mb + 16 + n) * DD + h * 16;

    v8f acc[2][4] = {};

    // uniform scalar lane-0 predicate so TDM is issued by exactly one wave
    const bool issuer = (__builtin_amdgcn_readfirstlane(threadIdx.x) < 32);
    (void)issuer;

#if HAVE_TDM
    const unsigned int ldsBase =
        (unsigned int)(uintptr_t)(&ldsW[0]);
    const unsigned long long wgBase =
        (unsigned long long)(uintptr_t)(wb) + (unsigned long long)eBase * DD * 2ull;
    if (issuer) {
        tdm_stage_weights(ldsBase, wgBase);      // chunk 0 -> buffer 0
        wait_tensor0();
    }
    __syncthreads();
#else
    {
        const unsigned char* gsrc = (const unsigned char*)wb +
                                    (size_t)eBase * DD * 2;
#pragma unroll
        for (int i = 0; i < 4; ++i) {
            const int chunk = threadIdx.x + i * 256;   // 1024 x 16B
            const int row = chunk >> 4;
            const int c16 = chunk & 15;
            *(uint4*)(&ldsW[row * WROW + c16 * 16]) =
                *(const uint4*)(gsrc + (size_t)row * (DD * 2) + c16 * 16);
        }
        __syncthreads();
    }
#endif

    for (int kc = 0; kc < NKC; ++kc) {
        const int cur = kc & 1;
        // stage next chunk into the other buffer while computing
        if (kc + 1 < NKC) {
#if HAVE_TDM
            if (issuer) {
                tdm_stage_weights(
                    (unsigned int)(uintptr_t)(&ldsW[(cur ^ 1) * WBUF]),
                    (unsigned long long)(uintptr_t)(wb) +
                        ((unsigned long long)eBase * DD + (unsigned long long)(kc + 1) * KC) * 2ull);
            }
#else
            const unsigned char* gsrc = (const unsigned char*)wb +
                ((size_t)eBase * DD + (size_t)(kc + 1) * KC) * 2;
            unsigned char* ldst = &ldsW[(cur ^ 1) * WBUF];
#pragma unroll
            for (int i = 0; i < 4; ++i) {
                const int chunk = threadIdx.x + i * 256;
                const int row = chunk >> 4;
                const int c16 = chunk & 15;
                *(uint4*)(ldst + row * WROW + c16 * 16) =
                    *(const uint4*)(gsrc + (size_t)row * (DD * 2) + c16 * 16);
            }
#endif
        }

        const unsigned char* wcur = &ldsW[cur * WBUF];
#pragma unroll
        for (int j = 0; j < 4; ++j) {
            const int kg = kc * KC + j * 32;
            v16bf af0, af1;
            {
                const v8bf lo = *(const v8bf*)(pa0 + kg);
                const v8bf hi = *(const v8bf*)(pa0 + kg + 8);
#pragma unroll
                for (int i = 0; i < 8; ++i) { af0[i] = lo[i]; af0[i + 8] = hi[i]; }
            }
            {
                const v8bf lo = *(const v8bf*)(pa1 + kg);
                const v8bf hi = *(const v8bf*)(pa1 + kg + 8);
#pragma unroll
                for (int i = 0; i < 8; ++i) { af1[i] = lo[i]; af1[i + 8] = hi[i]; }
            }
#pragma unroll
            for (int s = 0; s < 4; ++s) {
                const v16bf bf = *(const v16bf*)(wcur + (s * 16 + n) * WROW +
                                                 j * 64 + h * 32);
                acc[0][s] = __builtin_amdgcn_wmma_f32_16x16x32_bf16(
                    false, af0, false, bf, (short)0, acc[0][s], false, false);
                acc[1][s] = __builtin_amdgcn_wmma_f32_16x16x32_bf16(
                    false, af1, false, bf, (short)0, acc[1][s], false, false);
            }
        }

#if HAVE_TDM
        if (issuer && (kc + 1 < NKC)) wait_tensor0();
#endif
        __syncthreads();
    }

    // Epilogue: bias (+softplus for delta rows); transposed coalesced stores.
#pragma unroll
    for (int f = 0; f < 2; ++f) {
        const int tBase = mb + f * 16 + h * 8;
        const int b  = tBase >> 13;
        const int tl = tBase & (LL - 1);
#pragma unroll
        for (int s = 0; s < 4; ++s) {
            const int e = eBase + s * 16 + n;
            if (e >= EE) continue;
            float bias;
            float* dst;
            bool act;
            if (e < DD) {
                bias = bd[e];  act = true;
                dst = deltaT + (((size_t)(b * DD + e)) << 13) + tl;
            } else if (e < DD + NN) {
                bias = bb[e - DD];  act = false;
                dst = BmT + (((size_t)(b * NN + (e - DD))) << 13) + tl;
            } else {
                bias = bc[e - DD - NN];  act = false;
                dst = CmT + (((size_t)(b * NN + (e - DD - NN))) << 13) + tl;
            }
            float v[8];
#pragma unroll
            for (int r = 0; r < 8; ++r) {
                float z = acc[f][s][r] + bias;
                v[r] = act ? softplusf(z) : z;
            }
            *(float4*)(dst)     = make_float4(v[0], v[1], v[2], v[3]);
            *(float4*)(dst + 4) = make_float4(v[4], v[5], v[6], v[7]);
        }
    }
}

// ---------------------------------------------------------------------------
// K4: sequential scan. Lane = (d, n): n = lane&15, two d per wave.
// Block = 256 threads = 16 consecutive d for one batch b. Grid = (64, 4).
// ---------------------------------------------------------------------------
__global__ __launch_bounds__(256)
void scan_kernel(const float* __restrict__ deltaT,
                 const float* __restrict__ xT,
                 const float* __restrict__ BmT,
                 const float* __restrict__ CmT,
                 const float* __restrict__ A_log,
                 const float* __restrict__ Dpar,
                 float* __restrict__ y) {
    __shared__ float ys[16][17];

    const int b    = blockIdx.y;
    const int d0   = blockIdx.x * 16;
    const int wave = threadIdx.x >> 5;
    const int lane = threadIdx.x & 31;
    const int hh   = lane >> 4;
    const int n    = lane & 15;
    const int dloc = wave * 2 + hh;
    const int d    = d0 + dloc;

    const float A_dn = -__expf(A_log[d * NN + n]);
    const float Dp   = Dpar[d];

    const float* pd = deltaT + (((size_t)(b * DD + d)) << 13);
    const float* px = xT     + (((size_t)(b * DD + d)) << 13);
    const float* pB = BmT    + (((size_t)(b * NN + n)) << 13);
    const float* pC = CmT    + (((size_t)(b * NN + n)) << 13);
    float* py = y + ((size_t)b << 13) * DD;

    float hs = 0.0f;

    float dv = pd[n];
    float xv = px[n];
    float4 Bq0 = *(const float4*)(pB + 0),  Bq1 = *(const float4*)(pB + 4);
    float4 Bq2 = *(const float4*)(pB + 8),  Bq3 = *(const float4*)(pB + 12);
    float4 Cq0 = *(const float4*)(pC + 0),  Cq1 = *(const float4*)(pC + 4);
    float4 Cq2 = *(const float4*)(pC + 8),  Cq3 = *(const float4*)(pC + 12);

    for (int t0 = 0; t0 < LL; t0 += 16) {
        const int t1 = t0 + 16;
        float dvN = 0.f, xvN = 0.f;
        float4 BN0{}, BN1{}, BN2{}, BN3{}, CN0{}, CN1{}, CN2{}, CN3{};
        if (t1 < LL) {
            dvN = pd[t1 + n];
            xvN = px[t1 + n];
            BN0 = *(const float4*)(pB + t1 + 0);  BN1 = *(const float4*)(pB + t1 + 4);
            BN2 = *(const float4*)(pB + t1 + 8);  BN3 = *(const float4*)(pB + t1 + 12);
            CN0 = *(const float4*)(pC + t1 + 0);  CN1 = *(const float4*)(pC + t1 + 4);
            CN2 = *(const float4*)(pC + t1 + 8);  CN3 = *(const float4*)(pC + t1 + 12);
        }

        float Ba[16], Ca[16];
        ((float4*)Ba)[0] = Bq0; ((float4*)Ba)[1] = Bq1;
        ((float4*)Ba)[2] = Bq2; ((float4*)Ba)[3] = Bq3;
        ((float4*)Ca)[0] = Cq0; ((float4*)Ca)[1] = Cq1;
        ((float4*)Ca)[2] = Cq2; ((float4*)Ca)[3] = Cq3;

#pragma unroll
        for (int j = 0; j < 16; ++j) {
            const float dt = __shfl(dv, j, 16);
            const float xt = __shfl(xv, j, 16);
            const float aa = __expf(fminf(dt * A_dn, 2.0f));
            const float bbx = fminf(fmaxf(dt * Ba[j], -2.0f), 2.0f);
            hs = fminf(fmaxf(fmaf(aa, hs, bbx * xt), -100.0f), 100.0f);
            float p = hs * Ca[j];
            p += __shfl_xor(p, 1);
            p += __shfl_xor(p, 2);
            p += __shfl_xor(p, 4);
            p += __shfl_xor(p, 8);
            if (n == j) ys[j][dloc] = p + xt * Dp;
        }
        __syncthreads();
        {
            const int tl2 = threadIdx.x >> 4;
            const int dl2 = threadIdx.x & 15;
            py[(size_t)(t0 + tl2) * DD + d0 + dl2] = ys[tl2][dl2];
        }
        __syncthreads();

        dv = dvN; xv = xvN;
        Bq0 = BN0; Bq1 = BN1; Bq2 = BN2; Bq3 = BN3;
        Cq0 = CN0; Cq1 = CN1; Cq2 = CN2; Cq3 = CN3;
    }
}

// ---------------------------------------------------------------------------
// Host launcher
// ---------------------------------------------------------------------------
extern "C" void kernel_launch(void* const* d_in, const int* in_sizes, int n_in,
                              void* d_out, int out_size, void* d_ws, size_t ws_size,
                              hipStream_t stream) {
    const float* x     = (const float*)d_in[0];
    const float* Wd    = (const float*)d_in[1];
    const float* bd    = (const float*)d_in[2];
    const float* Wb    = (const float*)d_in[3];
    const float* bb    = (const float*)d_in[4];
    const float* Wc    = (const float*)d_in[5];
    const float* bc    = (const float*)d_in[6];
    const float* A_log = (const float*)d_in[7];
    const float* Dpar  = (const float*)d_in[8];
    float* y = (float*)d_out;

    char* ws = (char*)d_ws;
    const size_t szXT = (size_t)BB * DD * LL * 4;          // 128 MB
    const size_t szDT = (size_t)BB * DD * LL * 4;          // 128 MB
    const size_t szXB = (size_t)BB * LL * DD * 2;          // 64 MB
    const size_t szWB = (size_t)EE_PAD * DD * 2;           // ~2.2 MB
    const size_t szBT = (size_t)BB * NN * LL * 4;          // 2 MB

    float*          xT     = (float*)(ws);
    float*          deltaT = (float*)(ws + szXT);
    unsigned short* xb     = (unsigned short*)(ws + szXT + szDT);
    unsigned short* wb     = (unsigned short*)(ws + szXT + szDT + szXB);
    float*          BmT    = (float*)(ws + szXT + szDT + szXB + szWB);
    float*          CmT    = (float*)(ws + szXT + szDT + szXB + szWB + szBT);

    transpose_convert<<<dim3(LL / 32, DD / 32, BB), 256, 0, stream>>>(x, xT, xb);
    convert_w<<<dim3((EE_PAD * DD) / 256), 256, 0, stream>>>(Wd, Wb, Wc, wb);
    gemm_proj<<<dim3(17, (BB * LL) / 256), 256, 0, stream>>>(
        xb, wb, bd, bb, bc, deltaT, BmT, CmT);
    scan_kernel<<<dim3(DD / 16, BB), 256, 0, stream>>>(
        deltaT, xT, BmT, CmT, A_log, Dpar, y);
}
